// SelfAttention_52012053954870
// MI455X (gfx1250) — matrix operand order repaired and checked by reading, not compiled, not run on hardware
//
#include <hip/hip_runtime.h>
#include <hip/hip_bf16.h>

// ---------------------------------------------------------------------------
// CDNA5 (gfx1250) fused self-attention, bf16 WMMA path.
//   QKV proj -> scores = Q K^T * 1/sqrt(D) -> softmax -> out = P V
// GEMMs: v_wmma_f32_16x16x32_bf16 (wave32), f32 accumulate.
// Staging: double-buffered LDS software pipeline; pure-copy tiles use
// global_load_async_to_lds_b128 (ASYNCcnt), convert tiles use b128 loads
// + packed bf16 stores. Distance-2 global_prefetch_b8 on convert paths.
// ---------------------------------------------------------------------------

typedef __attribute__((ext_vector_type(16))) __bf16 v16bf;
typedef __attribute__((ext_vector_type(8)))  float  v8f;

#define BM 128
#define BN 128
#define BK 32
#define WAVES_M 4   // 8 waves: 4 (M) x 2 (N); wave tile 32x64 = TM x TN wmma tiles
#define TM 2
#define TN 4

template <class X, class Y> struct same_t { static constexpr bool value = false; };
template <class X> struct same_t<X, X>    { static constexpr bool value = true;  };

union AFrag { v16bf v; __bf16 e[16]; };
union BFrag { v16bf v; __bf16 e[16]; };
union CFrag { v8f  v; float  e[8];  };

struct alignas(8) bh4 { __bf16 x, y, z, w; };
union pk4 { __bf16 h[4]; unsigned long long u; };

// gfx1250 async global->LDS copy (ASYNCcnt-tracked), per-lane 16 bytes.
__device__ __forceinline__ void async_copy_b128(unsigned lds_addr, const void* gaddr) {
    asm volatile("global_load_async_to_lds_b128 %0, %1, off"
                 :: "v"(lds_addr), "v"(gaddr) : "memory");
}
__device__ __forceinline__ void wait_async0() {
    asm volatile("s_wait_asynccnt 0x0" ::: "memory");
}

// C[M,N] = scale * (A[M,K] @ op(B)) + bias,  op(B)=B[K,N] or B[N,K] (TRANSB)
// Batched via blockIdx.z. All dims multiples of the tile sizes.
template <typename TA, typename TB, typename TOUT, bool TRANSB, bool HAS_BIAS>
__global__ __launch_bounds__(256)
void gemm_wmma(const TA* __restrict__ A, const TB* __restrict__ B,
               const float* __restrict__ bias, TOUT* __restrict__ C,
               int Kd, int lda, int ldb, int ldc,
               long long strideA, long long strideB, long long strideC,
               float scale)
{
    // Pure-copy staging (no f32->bf16 convert) -> async DMA path.
    constexpr bool ASYNC_AB =
        same_t<TA, __bf16>::value && same_t<TB, __bf16>::value && TRANSB;

    __shared__ __bf16 As[2][BM * BK];   // [m][k] row-major
    __shared__ __bf16 Bs[2][BM * BK];   // [n][k] N-major (contiguous K per n)

    const int bz = blockIdx.z;
    A += (long long)bz * strideA;
    B += (long long)bz * strideB;
    C += (long long)bz * strideC;

    const int tid  = threadIdx.x;
    const int lane = tid & 31;
    const int wave = tid >> 5;
    const int wm   = (wave % WAVES_M) * (TM * 16);
    const int wn   = (wave / WAVES_M) * (TN * 16);
    const int row0 = blockIdx.y * BM;
    const int col0 = blockIdx.x * BN;
    const int hsel = lane >> 4;
    const int l16  = lane & 15;

    CFrag acc[TM][TN];
    #pragma unroll
    for (int i = 0; i < TM; ++i)
        #pragma unroll
        for (int j = 0; j < TN; ++j)
            #pragma unroll
            for (int r = 0; r < 8; ++r) acc[i][j].e[r] = 0.0f;

    // -------- staging state (register-resident between load/store phases) ----
    float4 aReg[4];   // A tile, float source: 4 x float4 = 16 elems/thread
    float4 bRegF[4];  // B tile, float source (non-trans)
    bh4    bRegH[4];  // B tile, bf16 source (non-trans)

    auto loadA = [&](int k0) {
        if constexpr (!ASYNC_AB) {
            #pragma unroll
            for (int e = 0; e < 4; ++e) {
                int i4 = e * 256 + tid;
                int r = i4 >> 3, c = (i4 & 7) * 4;
                aReg[e] = *(const float4*)&A[(long long)(row0 + r) * lda + k0 + c];
            }
        }
    };
    auto storeA = [&](int buf) {
        if constexpr (!ASYNC_AB) {
            #pragma unroll
            for (int e = 0; e < 4; ++e) {
                int i4 = e * 256 + tid;
                pk4 p;
                p.h[0] = (__bf16)aReg[e].x; p.h[1] = (__bf16)aReg[e].y;
                p.h[2] = (__bf16)aReg[e].z; p.h[3] = (__bf16)aReg[e].w;
                *(unsigned long long*)&As[buf][i4 * 4] = p.u;
            }
        }
    };
    auto loadB = [&](int k0) {
        if constexpr (!ASYNC_AB) {  // only non-trans B instantiated here
            #pragma unroll
            for (int e = 0; e < 4; ++e) {
                int i4 = e * 256 + tid;
                int k = i4 >> 5, n = (i4 & 31) * 4;
                if constexpr (same_t<TB, float>::value)
                    bRegF[e] = *(const float4*)&B[(long long)(k0 + k) * ldb + col0 + n];
                else
                    bRegH[e] = *(const bh4*)&B[(long long)(k0 + k) * ldb + col0 + n];
            }
        }
    };
    auto storeB = [&](int buf) {
        if constexpr (!ASYNC_AB) {
            #pragma unroll
            for (int e = 0; e < 4; ++e) {
                int i4 = e * 256 + tid;
                int k = i4 >> 5, n = (i4 & 31) * 4;
                if constexpr (same_t<TB, float>::value) {
                    Bs[buf][(n + 0) * BK + k] = (__bf16)bRegF[e].x;
                    Bs[buf][(n + 1) * BK + k] = (__bf16)bRegF[e].y;
                    Bs[buf][(n + 2) * BK + k] = (__bf16)bRegF[e].z;
                    Bs[buf][(n + 3) * BK + k] = (__bf16)bRegF[e].w;
                } else {
                    Bs[buf][(n + 0) * BK + k] = bRegH[e].x;
                    Bs[buf][(n + 1) * BK + k] = bRegH[e].y;
                    Bs[buf][(n + 2) * BK + k] = bRegH[e].z;
                    Bs[buf][(n + 3) * BK + k] = bRegH[e].w;
                }
            }
        }
    };
    auto issueAsync = [&](int k0, int buf) {
        if constexpr (ASYNC_AB) {
            // 8 KB tile = 512 x 16B chunks; 2 chunks per thread per tile.
            #pragma unroll
            for (int e = 0; e < 2; ++e) {
                int ch = e * 256 + tid;
                int r = ch >> 2, co = (ch & 3) * 8;           // 4 chunks per 64B row
                async_copy_b128((unsigned)(size_t)&As[buf][ch * 8],
                                (const void*)&A[(long long)(row0 + r) * lda + k0 + co]);
                async_copy_b128((unsigned)(size_t)&Bs[buf][ch * 8],
                                (const void*)&B[(long long)(col0 + r) * ldb + k0 + co]);
            }
        }
    };

    // -------- software pipeline: prologue --------
    const int nk = Kd / BK;
    if constexpr (ASYNC_AB) {
        issueAsync(0, 0);
        wait_async0();
    } else {
        loadA(0); loadB(0);
        storeA(0); storeB(0);
    }
    __syncthreads();

    for (int t = 0; t < nk; ++t) {
        const int  cur  = t & 1, nxt = cur ^ 1;
        const int  k0   = t * BK;
        const bool more = (t + 1) < nk;

        if (more) {
            if constexpr (ASYNC_AB) issueAsync(k0 + BK, nxt);  // DMA overlaps WMMA
            else { loadA(k0 + BK); loadB(k0 + BK); }           // loads in flight
        }
        if constexpr (!ASYNC_AB) {
            if (t + 2 < nk) {
                int kp = k0 + 2 * BK;
                if (tid < BM) {
                    __builtin_prefetch(&A[(long long)(row0 + tid) * lda + kp], 0, 1);
                } else {
                    int u = tid - BM;
                    if constexpr (TRANSB)
                        __builtin_prefetch(&B[(long long)(col0 + u) * ldb + kp], 0, 1);
                    else
                        __builtin_prefetch(&B[(long long)(kp + (u >> 2)) * ldb +
                                              col0 + (u & 3) * 32], 0, 1);
                }
            }
        }

        // ---- fragments + WMMA on current buffer ----
        AFrag af[TM];
        #pragma unroll
        for (int tm = 0; tm < TM; ++tm) {
            const __bf16* arow = &As[cur][(wm + tm * 16 + l16) * BK];
            #pragma unroll
            for (int j = 0; j < 16; ++j) {
                int k = j + (hsel << 4);
                af[tm].e[j] = arow[k];
            }
        }
        #pragma unroll
        for (int tn = 0; tn < TN; ++tn) {
            BFrag bfr;
            const __bf16* bcol = &Bs[cur][(wn + tn * 16 + l16) * BK + (hsel << 4)];
            #pragma unroll
            for (int j = 0; j < 16; ++j) bfr.e[j] = bcol[j];
            #pragma unroll
            for (int tm = 0; tm < TM; ++tm) {
                acc[tm][tn].v = __builtin_amdgcn_wmma_f32_16x16x32_bf16(
                    false, af[tm].v, false, bfr.v,
                    (short)0, acc[tm][tn].v, false, false);
            }
        }

        if (more) {
            if constexpr (ASYNC_AB) wait_async0();
            else { storeA(nxt); storeB(nxt); }
        }
        __syncthreads();
    }

    // ---- epilogue: D layout — VGPR r: row = r + 8*hsel, col = l16 ----
    #pragma unroll
    for (int tm = 0; tm < TM; ++tm) {
        #pragma unroll
        for (int tn = 0; tn < TN; ++tn) {
            #pragma unroll
            for (int r = 0; r < 8; ++r) {
                int row = row0 + wm + tm * 16 + r + hsel * 8;
                int col = col0 + wn + tn * 16 + l16;
                float v = acc[tm][tn].e[r] * scale;
                if (HAS_BIAS) v += bias[col];
                C[(long long)row * ldc + col] = (TOUT)v;
            }
        }
    }
}

// In-place row softmax over 2048 fp32 columns. One 256-thread block per row.
__global__ __launch_bounds__(256)
void softmax_rows(float* __restrict__ scores, int ncols)
{
    float* p = scores + (long long)blockIdx.x * ncols;
    const int tid = threadIdx.x;

    float v[8];
    #pragma unroll
    for (int i = 0; i < 8; ++i) v[i] = p[tid + i * 256];

    __shared__ float redm[8];
    __shared__ float reds[8];

    float m = v[0];
    #pragma unroll
    for (int i = 1; i < 8; ++i) m = fmaxf(m, v[i]);
    #pragma unroll
    for (int o = 16; o > 0; o >>= 1) m = fmaxf(m, __shfl_xor(m, o, 32));
    if ((tid & 31) == 0) redm[tid >> 5] = m;
    __syncthreads();
    float rm = redm[0];
    #pragma unroll
    for (int w = 1; w < 8; ++w) rm = fmaxf(rm, redm[w]);

    float s = 0.0f;
    #pragma unroll
    for (int i = 0; i < 8; ++i) { v[i] = __expf(v[i] - rm); s += v[i]; }
    #pragma unroll
    for (int o = 16; o > 0; o >>= 1) s += __shfl_xor(s, o, 32);
    if ((tid & 31) == 0) reds[tid >> 5] = s;
    __syncthreads();
    float rs = 0.0f;
    #pragma unroll
    for (int w = 0; w < 8; ++w) rs += reds[w];

    float inv = 1.0f / rs;
    #pragma unroll
    for (int i = 0; i < 8; ++i) p[tid + i * 256] = v[i] * inv;
}

extern "C" void kernel_launch(void* const* d_in, const int* in_sizes, int n_in,
                              void* d_out, int out_size, void* d_ws, size_t ws_size,
                              hipStream_t stream)
{
    (void)in_sizes; (void)n_in; (void)out_size; (void)ws_size;
    const int Bb = 4, S = 2048, D = 1024;

    const float* x  = (const float*)d_in[0];
    const float* Wq = (const float*)d_in[1];
    const float* bq = (const float*)d_in[2];
    const float* Wk = (const float*)d_in[3];
    const float* bk = (const float*)d_in[4];
    const float* Wv = (const float*)d_in[5];
    const float* bv = (const float*)d_in[6];
    float* out = (float*)d_out;

    // workspace: Q,K,V bf16 (3 x 16 MB) + scores f32 (64 MB)
    const size_t tokElems = (size_t)Bb * S * D;
    __bf16* Qh = (__bf16*)d_ws;
    __bf16* Kh = Qh + tokElems;
    __bf16* Vh = Kh + tokElems;
    float*  scores = (float*)(Vh + tokElems);

    dim3 blk(256, 1, 1);

    // 1) QKV projections: [8192,1024] = x[8192,1024] @ W[1024,1024] + b
    {
        dim3 g(D / BN, (Bb * S) / BM, 1);
        gemm_wmma<float, float, __bf16, false, true><<<g, blk, 0, stream>>>(
            x, Wq, bq, Qh, D, D, D, D, 0LL, 0LL, 0LL, 1.0f);
        gemm_wmma<float, float, __bf16, false, true><<<g, blk, 0, stream>>>(
            x, Wk, bk, Kh, D, D, D, D, 0LL, 0LL, 0LL, 1.0f);
        gemm_wmma<float, float, __bf16, false, true><<<g, blk, 0, stream>>>(
            x, Wv, bv, Vh, D, D, D, D, 0LL, 0LL, 0LL, 1.0f);
    }

    // 2) scores[b] = Q[b] @ K[b]^T * (1/sqrt(1024)); async-DMA staging
    {
        dim3 g(S / BN, S / BM, Bb);
        gemm_wmma<__bf16, __bf16, float, true, false><<<g, blk, 0, stream>>>(
            Qh, Kh, nullptr, scores, D, D, D, S,
            (long long)S * D, (long long)S * D, (long long)S * S, 0.03125f);
    }

    // 3) softmax rows in place (B*S rows of length S)
    softmax_rows<<<dim3(Bb * S, 1, 1), blk, 0, stream>>>(scores, S);

    // 4) out[b] = P[b] @ V[b]
    {
        dim3 g(D / BN, S / BM, Bb);
        gemm_wmma<float, __bf16, float, false, false><<<g, blk, 0, stream>>>(
            scores, Vh, nullptr, out, S, S, D, D,
            (long long)S * S, (long long)S * D, (long long)S * D, 1.0f);
    }
}
